// SOMVAE_18382460027423
// MI455X (gfx1250) — matrix operand order repaired and checked by reading, not compiled, not run on hardware
//
#include <hip/hip_runtime.h>
#include <hip/hip_bf16.h>

// ---------------------------------------------------------------------------
// SOM-VAE forward for MI455X (gfx1250, wave32, WMMA + async-LDS).
// B=16384, D_IN=2048, L=512, K=256 (16x16 SOM grid).
// ---------------------------------------------------------------------------

typedef __attribute__((ext_vector_type(16))) __bf16 v16bf;
typedef __attribute__((ext_vector_type(4)))  __bf16 v4bf;
typedef __attribute__((ext_vector_type(8)))  float  v8f;

#define NB     16384
#define DIN    2048
#define LAT    512
#define NEMB   256

// d_out flat offsets (floats), reference return order:
// (x_e, x_q, z_e, z_q, z_q_neighbors, k, z_dist_flat)
#define OFF_XE   ((size_t)0)
#define OFF_XQ   (OFF_XE  + (size_t)NB * DIN)
#define OFF_ZE   (OFF_XQ  + (size_t)NB * DIN)
#define OFF_ZQ   (OFF_ZE  + (size_t)NB * LAT)
#define OFF_ZQN  (OFF_ZQ  + (size_t)NB * LAT)
#define OFF_K    (OFF_ZQN + (size_t)NB * 5 * LAT)
#define OFF_DIST (OFF_K   + (size_t)NB)

#define TILE_STR 40      // LDS row stride in bf16 elements (32 + 8 pad)
#define USE_ASYNC_B 1    // stage B tiles with global_load_async_to_lds_b128

union FragCvt { v16bf v; float4 f[2]; };

__device__ __forceinline__ void wait_async0() {
#if defined(__has_builtin) && __has_builtin(__builtin_amdgcn_s_wait_asynccnt)
  __builtin_amdgcn_s_wait_asynccnt(0);
#else
  asm volatile("s_wait_asynccnt 0x0" ::: "memory");
#endif
}

// ---------------------------------------------------------------------------
// GEMM: C[M,N] = A[M,Kdim] @ B + epilogue, where Bt is bf16 [N][Kdim]
// (pre-transposed weight / emb_flat), A is f32.
//   EPI=0: C = acc + colv[n]
//   EPI=1: C = rowv[m] - 2*acc + colv[n]   (squared-distance expansion)
// Block: 256 threads (8 waves). Tile 128x128, K-step 32, double-buffered LDS.
// Wave (4x2 grid) owns 32 rows x 64 cols = 2x4 WMMA 16x16 tiles.
// ---------------------------------------------------------------------------
template<int EPI>
__global__ __launch_bounds__(256) void gemm_bf16_wmma(
    const float* __restrict__ A, const __bf16* __restrict__ Bt,
    const float* __restrict__ colv, const float* __restrict__ rowv,
    float* __restrict__ C, int Kdim, int N)
{
  __shared__ __bf16 As[2][128 * TILE_STR];
  __shared__ __bf16 Bs[2][128 * TILE_STR];

  const int tid   = threadIdx.x;
  const int lane  = tid & 31;
  const int wave  = tid >> 5;
  const int wm    = (wave & 3) * 32;   // wave's row offset inside 128-tile
  const int wn    = (wave >> 2) * 64;  // wave's col offset inside 128-tile
  const int bm    = blockIdx.y * 128;
  const int bn    = blockIdx.x * 128;
  const int lhalf = lane >> 4;         // 0: lanes 0-15, 1: lanes 16-31
  const int lrow  = lane & 15;

  // A-stage: 128x32 f32 tile, 4 float4 per thread.
  auto stageA_load = [&](int kt, float4* reg) {
    const int k0 = kt << 5;
    #pragma unroll
    for (int i = 0; i < 4; ++i) {
      const int idx = tid + i * 256;        // float4 slot 0..1023
      const int row = idx >> 3;
      const int c4  = (idx & 7) << 2;
      reg[i] = *(const float4*)(A + (size_t)(bm + row) * Kdim + k0 + c4);
    }
  };
  auto stageA_store = [&](int buf, const float4* reg) {
    #pragma unroll
    for (int i = 0; i < 4; ++i) {
      const int idx = tid + i * 256;
      const int row = idx >> 3;
      const int c4  = (idx & 7) << 2;
      v4bf o;
      o[0] = (__bf16)reg[i].x; o[1] = (__bf16)reg[i].y;
      o[2] = (__bf16)reg[i].z; o[3] = (__bf16)reg[i].w;
      *(v4bf*)(&As[buf][row * TILE_STR + c4]) = o;   // ds_store_b64
    }
  };
  // B-stage: 128x32 bf16 tile = 512 x 16B chunks, 2 per thread.
  auto stageB = [&](int kt, int buf) {
    const int k0 = kt << 5;
    #pragma unroll
    for (int i = 0; i < 2; ++i) {
      const int idx = tid + i * 256;        // chunk 0..511
      const int row = idx >> 2;
      const int c8  = (idx & 3) << 3;       // bf16 offset 0,8,16,24
      const __bf16* g = Bt + (size_t)(bn + row) * Kdim + k0 + c8;
      __bf16* l = &Bs[buf][row * TILE_STR + c8];
#if USE_ASYNC_B
      asm volatile("global_load_async_to_lds_b128 %0, %1, off"
                   :: "v"((unsigned int)(uintptr_t)l),
                      "v"((unsigned long long)(uintptr_t)g)
                   : "memory");
#else
      *(uint4*)l = *(const uint4*)g;
#endif
    }
  };

  v8f acc[2][4];
  #pragma unroll
  for (int mt = 0; mt < 2; ++mt)
    #pragma unroll
    for (int nt = 0; nt < 4; ++nt) { v8f z = {}; acc[mt][nt] = z; }

  const int ksteps = Kdim >> 5;
  float4 aReg[4];

  // prologue: stage k-step 0 into buffer 0
  stageA_load(0, aReg);
  stageB(0, 0);
  stageA_store(0, aReg);
#if USE_ASYNC_B
  wait_async0();
#endif
  __syncthreads();

  for (int kt = 0; kt < ksteps; ++kt) {
    const int cur  = kt & 1;
    const int nxt  = cur ^ 1;
    const bool more = (kt + 1 < ksteps);

    if (more) {           // issue next-step global traffic early
      stageA_load(kt + 1, aReg);
      stageB(kt + 1, nxt);
    }

    // fragments per documented CDNA5 bf16 layouts
    v16bf afrag[2], bfrag[4];
    #pragma unroll
    for (int mt = 0; mt < 2; ++mt) {
      // A 16x32: lanes 0-15: K0-7 + K16-23; lanes 16-31: K8-15 + K24-31
      const __bf16* p = &As[cur][(wm + mt * 16 + lrow) * TILE_STR + lhalf * 8];
      FragCvt fc;
      fc.f[0] = *(const float4*)(p);
      fc.f[1] = *(const float4*)(p + 16);
      afrag[mt] = fc.v;
    }
    #pragma unroll
    for (int nt = 0; nt < 4; ++nt) {
      // B 32x16 (col per lane): lanes 0-15: K0-7 + K16-23; lanes 16-31: K8-15 + K24-31
      const __bf16* p = &Bs[cur][(wn + nt * 16 + lrow) * TILE_STR + lhalf * 8];
      FragCvt fc;
      fc.f[0] = *(const float4*)(p);
      fc.f[1] = *(const float4*)(p + 16);
      bfrag[nt] = fc.v;
    }

    #pragma unroll
    for (int mt = 0; mt < 2; ++mt)
      #pragma unroll
      for (int nt = 0; nt < 4; ++nt)
        acc[mt][nt] = __builtin_amdgcn_wmma_f32_16x16x32_bf16(
            false, afrag[mt], false, bfrag[nt],
            (short)0, acc[mt][nt], false, false);

    if (more) {
      stageA_store(nxt, aReg);
#if USE_ASYNC_B
      wait_async0();
#endif
      __syncthreads();
    }
  }

  // epilogue: C/D layout = lanes 0-15: (M=j, N=lrow); 16-31: (M=j+8, N=lrow)
  #pragma unroll
  for (int mt = 0; mt < 2; ++mt) {
    #pragma unroll
    for (int nt = 0; nt < 4; ++nt) {
      const int col = bn + wn + nt * 16 + lrow;
      #pragma unroll
      for (int j = 0; j < 8; ++j) {
        const int row = bm + wm + mt * 16 + lhalf * 8 + j;
        float v = acc[mt][nt][j];
        if (EPI == 0) v += colv[col];
        else          v = rowv[row] - 2.0f * v + colv[col];
        C[(size_t)row * N + col] = v;
      }
    }
  }
}

// ---------------------------------------------------------------------------
// Weight prep: f32 [R][C] -> bf16 [C][R] (LDS-tiled transpose + convert).
// ---------------------------------------------------------------------------
__global__ __launch_bounds__(256) void transpose_cvt_bf16(
    const float* __restrict__ in, __bf16* __restrict__ out, int R, int C)
{
  __shared__ float t[32][33];
  const int c0 = blockIdx.x * 32;
  const int r0 = blockIdx.y * 32;
  #pragma unroll
  for (int j = 0; j < 4; ++j)
    t[threadIdx.y + j * 8][threadIdx.x] =
        in[(size_t)(r0 + threadIdx.y + j * 8) * C + c0 + threadIdx.x];
  __syncthreads();
  #pragma unroll
  for (int j = 0; j < 4; ++j)
    out[(size_t)(c0 + threadIdx.y + j * 8) * R + r0 + threadIdx.x] =
        (__bf16)t[threadIdx.x][threadIdx.y + j * 8];
}

// f32 -> bf16 straight convert (embeddings, already [N][K] layout).
__global__ __launch_bounds__(256) void cvt_bf16(
    const float* __restrict__ in, __bf16* __restrict__ out, int n4)
{
  const int i = blockIdx.x * blockDim.x + threadIdx.x;
  if (i < n4) {
    const float4 v = *(const float4*)(in + (size_t)i * 4);
    v4bf o;
    o[0] = (__bf16)v.x; o[1] = (__bf16)v.y; o[2] = (__bf16)v.z; o[3] = (__bf16)v.w;
    *(v4bf*)(out + (size_t)i * 4) = o;
  }
}

// ---------------------------------------------------------------------------
// Row sum of squares: one wave per row (ncols multiple of 128).
// ---------------------------------------------------------------------------
__global__ __launch_bounds__(256) void rowsumsq(
    const float* __restrict__ src, float* __restrict__ dst, int ncols)
{
  const int lane = threadIdx.x & 31;
  const int row  = (int)((blockIdx.x * blockDim.x + threadIdx.x) >> 5);
  const float4* p = (const float4*)(src + (size_t)row * ncols);
  const int nf4 = ncols >> 2;
  float s = 0.0f;
  for (int c = lane; c < nf4; c += 32) {
    const float4 v = p[c];
    s += v.x * v.x + v.y * v.y + v.z * v.z + v.w * v.w;
  }
  #pragma unroll
  for (int off = 16; off; off >>= 1) s += __shfl_xor(s, off, 32);
  if (lane == 0) dst[row] = s;
}

// ---------------------------------------------------------------------------
// Per-row argmin over 256 distances (first-min tie-break, like jnp.argmin)
// + gather z_q and SOM grid neighbors. One wave per row.
// Reproduces reference bug: z_q_right is all zeros.
// ---------------------------------------------------------------------------
__global__ __launch_bounds__(256) void argmin_gather(
    const float* __restrict__ dist,   // [B, 256]
    const float* __restrict__ emb,    // [256, 512]
    float* __restrict__ out_k,        // [B] (float-coded index)
    float* __restrict__ out_zq,       // [B, 512]
    float* __restrict__ out_zqn)      // [B, 5, 512]
{
  const int lane = threadIdx.x & 31;
  const int row  = (int)((blockIdx.x * blockDim.x + threadIdx.x) >> 5);

  const float* d = dist + (size_t)row * NEMB;
  float bestv = 3.402823466e38f;
  int   besti = 0x7fffffff;
  #pragma unroll
  for (int c = 0; c < 2; ++c) {
    const int base = lane * 8 + c * 4;
    const float4 v = *(const float4*)(d + base);
    if (v.x < bestv) { bestv = v.x; besti = base;     }
    if (v.y < bestv) { bestv = v.y; besti = base + 1; }
    if (v.z < bestv) { bestv = v.z; besti = base + 2; }
    if (v.w < bestv) { bestv = v.w; besti = base + 3; }
  }
  #pragma unroll
  for (int off = 16; off; off >>= 1) {
    const float ov = __shfl_xor(bestv, off, 32);
    const int   oi = __shfl_xor(besti, off, 32);
    if (ov < bestv || (ov == bestv && oi < besti)) { bestv = ov; besti = oi; }
  }

  const int k  = besti;
  const int k1 = k >> 4;
  const int k2 = k & 15;
  if (lane == 0) out_k[row] = (float)k;

  const bool has_up = (k1 < 15), has_dn = (k1 > 0), has_lf = (k2 > 0);
  const int ku = has_up ? (k + 16) : k;   // clamp so loads stay in-bounds
  const int kd = has_dn ? (k - 16) : k;
  const int kl = has_lf ? (k - 1)  : k;

  const float4* zq = (const float4*)(emb + (size_t)k  * LAT);
  const float4* up = (const float4*)(emb + (size_t)ku * LAT);
  const float4* dn = (const float4*)(emb + (size_t)kd * LAT);
  const float4* lf = (const float4*)(emb + (size_t)kl * LAT);

  float4* ozq = (float4*)(out_zq  + (size_t)row * LAT);
  float4* ozn = (float4*)(out_zqn + (size_t)row * 5 * LAT);
  const float4 zero = {0.0f, 0.0f, 0.0f, 0.0f};

  #pragma unroll
  for (int c = 0; c < 4; ++c) {
    const int f4 = lane + c * 32;   // 0..127 float4 slots of a 512-vector
    const float4 v  = zq[f4];
    const float4 vu = up[f4];
    const float4 vd = dn[f4];
    const float4 vl = lf[f4];
    ozq[f4]       = v;
    ozn[f4]       = v;                    // slot 0: z_q
    ozn[128 + f4] = has_up ? vu : zero;   // slot 1: up
    ozn[256 + f4] = has_dn ? vd : zero;   // slot 2: down
    ozn[384 + f4] = zero;                 // slot 3: right (reference `==` bug)
    ozn[512 + f4] = has_lf ? vl : zero;   // slot 4: left
  }
}

// ---------------------------------------------------------------------------
extern "C" void kernel_launch(void* const* d_in, const int* in_sizes, int n_in,
                              void* d_out, int out_size, void* d_ws, size_t ws_size,
                              hipStream_t stream) {
  (void)in_sizes; (void)n_in; (void)out_size; (void)ws_size;

  const float* x       = (const float*)d_in[0];
  const float* enc_w   = (const float*)d_in[1];
  const float* enc_b   = (const float*)d_in[2];
  const float* dec_q_w = (const float*)d_in[3];
  const float* dec_q_b = (const float*)d_in[4];
  const float* dec_e_w = (const float*)d_in[5];
  const float* dec_e_b = (const float*)d_in[6];
  const float* emb     = (const float*)d_in[7];

  float* out  = (float*)d_out;
  float* xe   = out + OFF_XE;
  float* xq   = out + OFF_XQ;
  float* ze   = out + OFF_ZE;
  float* zq   = out + OFF_ZQ;
  float* zqn  = out + OFF_ZQN;
  float* kk   = out + OFF_K;
  float* dist = out + OFF_DIST;

  // workspace: [zsq | esq | enc_wT | dec_q_wT | dec_e_wT | emb_bf]
  float*  zsq      = (float*)d_ws;                 // [16384]
  float*  esq      = zsq + NB;                     // [256]
  __bf16* enc_wT   = (__bf16*)(esq + NEMB);        // [512][2048]
  __bf16* dec_q_wT = enc_wT   + (size_t)LAT * DIN; // [2048][512]
  __bf16* dec_e_wT = dec_q_wT + (size_t)DIN * LAT; // [2048][512]
  __bf16* emb_bf   = dec_e_wT + (size_t)DIN * LAT; // [256][512]

  const dim3 blk(256);
  const dim3 tblk(32, 8);

  // 0. weight prep: bf16, pre-transposed to [N][K]
  transpose_cvt_bf16<<<dim3(LAT / 32, DIN / 32), tblk, 0, stream>>>(enc_w,   enc_wT,   DIN, LAT);
  transpose_cvt_bf16<<<dim3(DIN / 32, LAT / 32), tblk, 0, stream>>>(dec_q_w, dec_q_wT, LAT, DIN);
  transpose_cvt_bf16<<<dim3(DIN / 32, LAT / 32), tblk, 0, stream>>>(dec_e_w, dec_e_wT, LAT, DIN);
  cvt_bf16<<<(NEMB * LAT / 4 + 255) / 256, blk, 0, stream>>>(emb, emb_bf, NEMB * LAT / 4);

  // 1. z_e = x @ enc_w + enc_b                 (M=16384, K=2048, N=512)
  gemm_bf16_wmma<0><<<dim3(LAT / 128, NB / 128), blk, 0, stream>>>(
      x, enc_wT, enc_b, nullptr, ze, DIN, LAT);

  // 2. z_sq (rows of z_e), e_sq (rows of emb_flat)
  rowsumsq<<<NB / 8, blk, 0, stream>>>(ze, zsq, LAT);
  rowsumsq<<<NEMB / 8, blk, 0, stream>>>(emb, esq, LAT);

  // 3. z_dist = z_sq - 2 * (z_e @ emb^T) + e_sq  (M=16384, K=512, N=256)
  gemm_bf16_wmma<1><<<dim3(NEMB / 128, NB / 128), blk, 0, stream>>>(
      ze, emb_bf, esq, zsq, dist, LAT, NEMB);

  // 4. argmin + gather z_q / neighbors / k
  argmin_gather<<<NB / 8, blk, 0, stream>>>(dist, emb, kk, zq, zqn);

  // 5. x_q = z_q @ dec_q_w + dec_q_b           (M=16384, K=512, N=2048)
  gemm_bf16_wmma<0><<<dim3(DIN / 128, NB / 128), blk, 0, stream>>>(
      zq, dec_q_wT, dec_q_b, nullptr, xq, LAT, DIN);

  // 6. x_e = z_e @ dec_e_w + dec_e_b           (M=16384, K=512, N=2048)
  gemm_bf16_wmma<0><<<dim3(DIN / 128, NB / 128), blk, 0, stream>>>(
      ze, dec_e_wT, dec_e_b, nullptr, xe, LAT, DIN);
}
